// DotProductAttention_38379827757143
// MI455X (gfx1250) — matrix operand order repaired and checked by reading, not compiled, not run on hardware
//
#include <hip/hip_runtime.h>

typedef __attribute__((ext_vector_type(16))) _Float16 v16h;
typedef __attribute__((ext_vector_type(8)))  _Float16 v8h;
typedef __attribute__((ext_vector_type(2)))  _Float16 h2;
typedef __attribute__((ext_vector_type(8)))  float    v8f;
typedef __attribute__((ext_vector_type(4)))  float    f32x4;

#define B_  32
#define Q_  1024
#define K_  1024
#define D_  128
#define V_  128
#define KC  32          // key-chunk = WMMA K for f16
#define NW  4           // waves per block
#define QT  16          // query rows per wave
#define KLP (D_ + 8)    // klds padded row stride (halfs): 272 B -> conflict-free
#define VLP (KC + 8)    // vlds padded row stride (halfs): 80 B  -> conflict-free
#define PLP (KC + 8)    // plds padded row stride (halfs)

__global__ __launch_bounds__(NW * 32) void
fa_fwd_kernel(const float* __restrict__ q, const float* __restrict__ kk,
              const float* __restrict__ vv, const long long* __restrict__ vlen,
              float* __restrict__ out)
{
    __shared__ __align__(16) _Float16 klds[KC][KLP];      // K chunk f16 [k][d], 8.5 KB
    __shared__ __align__(16) _Float16 vlds[V_][VLP];      // V chunk f16 [n][k], 10 KB
    __shared__ __align__(16) _Float16 plds[NW][QT][PLP];  // P transpose, 5 KB

    const int tid  = threadIdx.x;
    const int wv   = tid >> 5;
    const int lane = tid & 31;
    const int r    = lane & 15;   // row / column index within 16-wide tile
    const int hh   = lane >> 4;   // lane half

    const int b  = blockIdx.x >> 4;                       // 16 q-blocks per batch
    const int qb = (blockIdx.x & 15) * (NW * QT) + wv * QT;

    const long long vl = vlen[b];
    const int nch = (int)((vl + (KC - 1)) / KC);          // skip fully-masked chunks

    const float scale = 0.088388347648318447f;            // 1/sqrt(128)

    // ---- Q A-fragments (16x32 f16), scale folded in; afrag[s] = dims [s*32, s*32+32)
    // A layout: lane holds row r; element e -> k = hh*8 + (e<8 ? e : 8+e)
    v16h afrag[4];
    {
        const float* qrow = q + ((size_t)b * Q_ + (qb + r)) * D_;
        #pragma unroll
        for (int s = 0; s < 4; ++s) {
            const float* p0 = qrow + s * 32 + hh * 8;
            const float* p1 = qrow + s * 32 + 16 + hh * 8;
            f32x4 x0 = *(const f32x4*)(p0);
            f32x4 x1 = *(const f32x4*)(p0 + 4);
            f32x4 y0 = *(const f32x4*)(p1);
            f32x4 y1 = *(const f32x4*)(p1 + 4);
            v16h a;
            #pragma unroll
            for (int e = 0; e < 4; ++e) {
                a[e]      = (_Float16)(x0[e] * scale);
                a[4 + e]  = (_Float16)(x1[e] * scale);
                a[8 + e]  = (_Float16)(y0[e] * scale);
                a[12 + e] = (_Float16)(y1[e] * scale);
            }
            afrag[s] = a;
        }
    }

    // constant all-ones B fragment: row-sum of P via one WMMA
    v16h ones;
    #pragma unroll
    for (int e = 0; e < 16; ++e) ones[e] = (_Float16)1.0f;

    v8f oacc[8];
    #pragma unroll
    for (int t = 0; t < 8; ++t) oacc[t] = (v8f){0.f,0.f,0.f,0.f,0.f,0.f,0.f,0.f};
    float mrun[8], lrun[8];                // row stats, row m = j + 8*hh
    #pragma unroll
    for (int j = 0; j < 8; ++j) { mrun[j] = -3.0e38f; lrun[j] = 0.f; }

    for (int c = 0; c < nch; ++c) {
        const int kc = c * KC;

        // =============== cooperative staging (once per block) ===============
        __syncthreads();
        {
            // --- K chunk -> klds[k][d] f16, row-major ---
            const int krow = tid >> 2;            // 0..31
            const int kcs  = (tid & 3) * 32;      // column start
            const float* kp = kk + ((size_t)b * K_ + (kc + krow)) * D_ + kcs;
            float ra[32];
            #pragma unroll
            for (int g = 0; g < 8; ++g) {
                f32x4 x = *(const f32x4*)(kp + g * 4);
                ra[g*4+0] = x[0]; ra[g*4+1] = x[1];
                ra[g*4+2] = x[2]; ra[g*4+3] = x[3];
            }
            #pragma unroll
            for (int g = 0; g < 4; ++g) {
                v8h hv;
                #pragma unroll
                for (int e = 0; e < 8; ++e) hv[e] = (_Float16)ra[g*8+e];
                *(v8h*)&klds[krow][kcs + g * 8] = hv;
            }

            // --- V chunk -> vlds[n][k] f16, transposed; pack row-pairs as b32 ---
            const int rp = tid >> 3;              // 0..15 : rows 2rp, 2rp+1
            const int cs = (tid & 7) * 16;        // column start
            const float* v0 = vv + ((size_t)b * K_ + (kc + 2 * rp)) * V_ + cs;
            float va[16], vb[16];
            #pragma unroll
            for (int g = 0; g < 4; ++g) {
                f32x4 xa = *(const f32x4*)(v0 + g * 4);
                f32x4 xb = *(const f32x4*)(v0 + V_ + g * 4);
                va[g*4+0]=xa[0]; va[g*4+1]=xa[1]; va[g*4+2]=xa[2]; va[g*4+3]=xa[3];
                vb[g*4+0]=xb[0]; vb[g*4+1]=xb[1]; vb[g*4+2]=xb[2]; vb[g*4+3]=xb[3];
            }
            #pragma unroll
            for (int e = 0; e < 16; ++e) {
                h2 p; p[0] = (_Float16)va[e]; p[1] = (_Float16)vb[e];
                *(h2*)&vlds[cs + e][2 * rp] = p;   // 4-byte aligned packed store
            }
        }
        __syncthreads();

        // prefetch next K/V chunks (global_prefetch_b8)
        if (c + 1 < nch) {
            __builtin_prefetch(kk + ((size_t)b * K_ + (kc + KC + lane)) * D_, 0, 1);
            __builtin_prefetch(vv + ((size_t)b * K_ + (kc + KC + lane)) * V_, 0, 1);
        }

        // ---- S = Q * K^T : two 16-column tiles, accumulate over D in 4 slices ----
        // B layout: lane holds column; element e -> k-dim = hh*8 + (e<8 ? e : 8+e)
        v8f c0 = {}, c1 = {};
        #pragma unroll
        for (int s = 0; s < 4; ++s) {
            const _Float16* k0 = &klds[r][s * 32 + hh * 8];
            v8h x0 = *(const v8h*)(k0);
            v8h x1 = *(const v8h*)(k0 + 16);
            v16h bf0 = __builtin_shufflevector(x0, x1, 0,1,2,3,4,5,6,7,8,9,10,11,12,13,14,15);
            c0 = __builtin_amdgcn_wmma_f32_16x16x32_f16(false, afrag[s], false, bf0,
                                                        (short)0, c0, false, false);
            const _Float16* k1 = &klds[16 + r][s * 32 + hh * 8];
            v8h y0 = *(const v8h*)(k1);
            v8h y1 = *(const v8h*)(k1 + 16);
            v16h bf1 = __builtin_shufflevector(y0, y1, 0,1,2,3,4,5,6,7,8,9,10,11,12,13,14,15);
            c1 = __builtin_amdgcn_wmma_f32_16x16x32_f16(false, afrag[s], false, bf1,
                                                        (short)0, c1, false, false);
        }

        // ---- mask (per-column, matches reference's -1e6 fill) ----
        const bool m0 = (long long)(kc + r)      < vl;
        const bool m1 = (long long)(kc + 16 + r) < vl;
        #pragma unroll
        for (int j = 0; j < 8; ++j) {
            if (!m0) c0[j] = -1.0e6f;
            if (!m1) c1[j] = -1.0e6f;
        }

        // ---- online softmax: max-reduce across 16 lanes, exp, write P ----
        float al[8];
        #pragma unroll
        for (int j = 0; j < 8; ++j) {
            float mx = fmaxf(c0[j], c1[j]);
            #pragma unroll
            for (int off = 8; off >= 1; off >>= 1)
                mx = fmaxf(mx, __shfl_xor(mx, off, 32));
            const float mn = fmaxf(mrun[j], mx);
            al[j] = __expf(mrun[j] - mn);
            const float p0 = __expf(c0[j] - mn);
            const float p1 = __expf(c1[j] - mn);
            plds[wv][j + 8 * hh][r]      = (_Float16)p0;   // write P transposed
            plds[wv][j + 8 * hh][16 + r] = (_Float16)p1;
            mrun[j] = mn;
            #pragma unroll
            for (int t = 0; t < 8; ++t) oacc[t][j] *= al[j];  // rescale running O
        }

        // ---- P as A-fragment (DS ops are in-order within a wave) ----
        v16h pf;
        {
            const _Float16* prow = &plds[wv][r][0];
            v8h lo = *(const v8h*)(prow + hh * 8);
            v8h hi = *(const v8h*)(prow + 16 + hh * 8);
            pf = __builtin_shufflevector(lo, hi, 0,1,2,3,4,5,6,7,8,9,10,11,12,13,14,15);
        }

        // ---- row sums of P on the tensor core: P(16x32) x ones(32x16) ----
        v8f sums = __builtin_amdgcn_wmma_f32_16x16x32_f16(false, pf, false, ones,
                                                          (short)0, (v8f){0.f,0.f,0.f,0.f,0.f,0.f,0.f,0.f},
                                                          false, false);
        #pragma unroll
        for (int j = 0; j < 8; ++j) lrun[j] = lrun[j] * al[j] + sums[j];

        // ---- O += P * Vchunk : 8 output column tiles ----
        #pragma unroll
        for (int t = 0; t < 8; ++t) {
            const _Float16* vrow = &vlds[t * 16 + r][hh * 8];
            v8h a0 = *(const v8h*)(vrow);
            v8h a1 = *(const v8h*)(vrow + 16);
            v16h vf = __builtin_shufflevector(a0, a1, 0,1,2,3,4,5,6,7,8,9,10,11,12,13,14,15);
            oacc[t] = __builtin_amdgcn_wmma_f32_16x16x32_f16(false, pf, false, vf,
                                                             (short)0, oacc[t], false, false);
        }
    }

    // ---- normalize and store: lane holds column t*16 + r of rows j + 8*hh ----
    #pragma unroll
    for (int j = 0; j < 8; ++j) {
        const float inv = 1.0f / lrun[j];
        float* orow = out + ((size_t)b * Q_ + (qb + j + 8 * hh)) * V_;
        #pragma unroll
        for (int t = 0; t < 8; ++t)
            orow[t * 16 + r] = oacc[t][j] * inv;
    }
}

extern "C" void kernel_launch(void* const* d_in, const int* in_sizes, int n_in,
                              void* d_out, int out_size, void* d_ws, size_t ws_size,
                              hipStream_t stream) {
    const float*     q  = (const float*)d_in[0];
    const float*     k  = (const float*)d_in[1];
    const float*     v  = (const float*)d_in[2];
    const long long* vl = (const long long*)d_in[3];
    float* out = (float*)d_out;

    dim3 grid(B_ * (Q_ / (NW * QT)));   // 32 * 16 = 512 blocks
    dim3 block(NW * 32);                // 4 waves
    fa_fwd_kernel<<<grid, block, 0, stream>>>(q, k, v, vl, out);

    (void)in_sizes; (void)n_in; (void)out_size; (void)d_ws; (void)ws_size;
}
